// MultiHeadAttentionLyaer_15393162789360
// MI455X (gfx1250) — matrix operand order repaired and checked by reading, not compiled, not run on hardware
//
#include <hip/hip_runtime.h>

// ---------------------------------------------------------------------------
// MHA for gfx1250 (CDNA5): bf16 WMMA (v_wmma_f32_16x16x32_bf16), f32 accum.
// B=2, S=2048, D=1024, H=16, dk=64.  Compute-bound (~69 GFLOP vs ~150 MB),
// so everything funnels through the 16x16x32 bf16 WMMA path; LDS tiles are
// double-buffered and staged via GLOBAL_LOAD_ASYNC_TO_LDS (ASYNCcnt).
// ---------------------------------------------------------------------------

typedef __attribute__((ext_vector_type(16))) __bf16       v16bf;
typedef __attribute__((ext_vector_type(8)))  float        v8f;
typedef __attribute__((ext_vector_type(4)))  unsigned int v4u;
typedef __attribute__((ext_vector_type(8)))  unsigned int v8u;

#define S_LEN  2048
#define DMODEL 1024
#define NHEAD  16
#define DK     64
#define MROWS  4096   // B*S

#if __has_builtin(__builtin_amdgcn_global_load_async_to_lds_b128) && \
    __has_builtin(__builtin_amdgcn_s_wait_asynccnt)
#define USE_ASYNC_LDS 1
#endif

#ifdef USE_ASYNC_LDS
// Builtin prototype (from hipcc diagnostic): first param is
// '__attribute__((vector_size(16))) int __device__ *' (AS1, non-const),
// second is the AS3 equivalent; args 2/3 are immediate offset / cpol.
typedef int v4i_vs __attribute__((vector_size(16)));
#define AS1P __attribute__((address_space(1))) v4i_vs*
#define AS3P __attribute__((address_space(3))) v4i_vs*
#endif

__device__ __forceinline__ unsigned short f2bf(float f) {
  unsigned int u = __builtin_bit_cast(unsigned int, f);
  unsigned int r = u + 0x7FFFu + ((u >> 16) & 1u);   // round-to-nearest-even
  return (unsigned short)(r >> 16);
}

__device__ __forceinline__ v8f wmma_bf16(v16bf a, v16bf b, v8f c) {
  // D = A(16x32 bf16) x B(32x16 bf16) + C(16x16 f32)
  return __builtin_amdgcn_wmma_f32_16x16x32_bf16(
      /*neg_a=*/false, a, /*neg_b=*/false, b,
      /*c_mod=*/(short)0, c, /*reuse_a=*/false, /*reuse_b=*/false);
}

// A fragment (16x32 bf16). Lane l holds row m=l&15; half h=l>>4 holds
// K chunk [16h,16h+16) contiguous (2 bf16 per VGPR).
__device__ __forceinline__ v16bf load_frag_a(const unsigned short* p, int ld, int lane) {
  const int m = lane & 15, hl = lane >> 4;
  const unsigned short* r = p + m * ld + hl * 16;
  v4u lo = *(const v4u*)(r);        // K = 16h .. 16h+7
  v4u hi = *(const v4u*)(r + 8);    // K = 16h+8 .. 16h+15
  v8u c;
  c[0] = lo[0]; c[1] = lo[1]; c[2] = lo[2]; c[3] = lo[3];
  c[4] = hi[0]; c[5] = hi[1]; c[6] = hi[2]; c[7] = hi[3];
  return __builtin_bit_cast(v16bf, c);
}

// B fragment (32x16 bf16). Lane l holds column n=l&15; half h=l>>4 holds
// K chunk [16h,16h+16) contiguous. p points at element [n0, k0] of a
// row-major matrix whose rows are B columns.
__device__ __forceinline__ v16bf load_frag_b(const unsigned short* p, int ld, int lane) {
  const int n = lane & 15, hl = lane >> 4;
  v8u c = *(const v8u*)(p + n * ld + hl * 16);
  return __builtin_bit_cast(v16bf, c);
}

// 16-byte global -> LDS copy: async-to-LDS (ASYNCcnt) when available.
__device__ __forceinline__ void cp16(unsigned short* ldst, const unsigned short* gsrc) {
#ifdef USE_ASYNC_LDS
  __builtin_amdgcn_global_load_async_to_lds_b128(
      (AS1P)const_cast<unsigned short*>(gsrc),
      (AS3P)ldst, 0, 0);
#else
  *(v4u*)ldst = *(const v4u*)gsrc;
#endif
}

// Drain this wave's staging traffic, then workgroup barrier.
__device__ __forceinline__ void wait_stage() {
#ifdef USE_ASYNC_LDS
  __builtin_amdgcn_s_wait_asynccnt(0);
#endif
  __syncthreads();
}

// ---------------------------------------------------------------------------
// fp32 -> bf16 conversion (vectorized, 4 elems/thread)
// ---------------------------------------------------------------------------
__global__ void cvt_bf16_kernel(const float* __restrict__ in,
                                unsigned short* __restrict__ out, int n4) {
  int i = blockIdx.x * blockDim.x + threadIdx.x;
  if (i < n4) {
    float4 f = ((const float4*)in)[i];
    unsigned long long pack =
        (unsigned long long)f2bf(f.x) |
        ((unsigned long long)f2bf(f.y) << 16) |
        ((unsigned long long)f2bf(f.z) << 32) |
        ((unsigned long long)f2bf(f.w) << 48);
    ((unsigned long long*)out)[i] = pack;
  }
}

// ---------------------------------------------------------------------------
// Y = X @ W^T + bias.  X:[M,1024] bf16, W:[1024,1024] bf16, both row-major.
// Block: 256 thr (8 waves), tile 256(M) x 64(N); wave tile 32x64 (8 accs,
// 8 WMMA / k-step, each B frag feeds 2 WMMAs). W k-tile (64x32 = 4 KB)
// double-buffered in LDS; stage of tile i+1 overlaps compute of tile i.
// mode 0: bf16 out, [B,H,S,dk];  mode 1: bf16 out, [B,H,dk,S] (for V);
// mode 2: f32 out, row-major [M,1024].
// ---------------------------------------------------------------------------
__global__ __launch_bounds__(256) void proj_gemm_kernel(
    const unsigned short* __restrict__ X,
    const unsigned short* __restrict__ W,
    const float* __restrict__ bias,
    unsigned short* __restrict__ Ybf,
    float* __restrict__ Yf,
    int mode) {
  __shared__ unsigned short lw[2][64 * 32];   // 2 x 4 KB W tiles

  const int lane = threadIdx.x & 31;
  const int wave = threadIdx.x >> 5;
  const int t    = threadIdx.x;
  const int m0 = blockIdx.y * 256 + wave * 32;
  const int n0 = blockIdx.x * 64;

  const v8f zero = {0.f, 0.f, 0.f, 0.f, 0.f, 0.f, 0.f, 0.f};
  v8f acc[8] = {zero, zero, zero, zero, zero, zero, zero, zero};

  const unsigned short* xrow0 = X + (size_t)m0 * DMODEL;
  const unsigned short* xrow1 = X + (size_t)(m0 + 16) * DMODEL;
  const int wr = t >> 2, wc = (t & 3) * 8;    // 64 rows x 4 chunks of 8

  // prologue: stage k-tile 0
  cp16(&lw[0][wr * 32 + wc], W + (size_t)(n0 + wr) * DMODEL + wc);

  for (int it = 0; it < DMODEL / 32; ++it) {
    const int k0 = it * 32;
    wait_stage();                              // tile (it&1) ready for all waves
    if (it + 1 < DMODEL / 32)                  // overlap: stage next tile
      cp16(&lw[(it + 1) & 1][wr * 32 + wc],
           W + (size_t)(n0 + wr) * DMODEL + (k0 + 32) + wc);

    if (k0 + 32 < DMODEL)
      __builtin_prefetch(xrow0 + k0 + 32, 0, 1);

    const v16bf a0 = load_frag_a(xrow0 + k0, DMODEL, lane);
    const v16bf a1 = load_frag_a(xrow1 + k0, DMODEL, lane);
    const unsigned short* wt = lw[it & 1];
#pragma unroll
    for (int nt = 0; nt < 4; ++nt) {
      const v16bf b = load_frag_b(wt + nt * 16 * 32, 32, lane);
      acc[nt]     = wmma_bf16(a0, b, acc[nt]);
      acc[4 + nt] = wmma_bf16(a1, b, acc[4 + nt]);
    }
  }

  // epilogue: C/D layout -> lane l: col n=l&15, rows m = 8*(l>>4)+v
  const int nl = lane & 15, hl = lane >> 4;
#pragma unroll
  for (int half = 0; half < 2; ++half) {
#pragma unroll
    for (int nt = 0; nt < 4; ++nt) {
      const int n = n0 + nt * 16 + nl;
      const float bvv = bias[n];
      const v8f av = acc[half * 4 + nt];
#pragma unroll
      for (int v = 0; v < 8; ++v) {
        const int m = m0 + half * 16 + hl * 8 + v;
        const float val = av[v] + bvv;
        if (mode == 2) {
          Yf[(size_t)m * DMODEL + n] = val;
        } else {
          const int b = m >> 11, s = m & (S_LEN - 1);
          const int hd = n >> 6, d = n & (DK - 1);
          if (mode == 0)
            Ybf[(((size_t)(b * NHEAD + hd)) * S_LEN + s) * DK + d] = f2bf(val);
          else
            Ybf[(((size_t)(b * NHEAD + hd)) * DK + d) * S_LEN + s] = f2bf(val);
        }
      }
    }
  }
}

// ---------------------------------------------------------------------------
// Flash attention. Block = 8 waves = 128 query rows of one (b, head); wave
// owns 16 rows. All waves consume the SAME 32-key K/V tiles, so tiles are
// cooperatively staged into double-buffered LDS (K: 32x64, V^T: 64x32).
// Per tile: 4 WMMAs QK^T, online softmax (width-16 xor reductions), P tile
// round-trips per-wave LDS (C-layout -> A-layout), 4 WMMAs P.V.
// ---------------------------------------------------------------------------
__global__ __launch_bounds__(256) void flash_attn_kernel(
    const unsigned short* __restrict__ Qh,   // [B,H,S,64]
    const unsigned short* __restrict__ Kh,   // [B,H,S,64]
    const unsigned short* __restrict__ Vt,   // [B,H,64,S]
    const int* __restrict__ mask,            // [B,1,S,S]
    unsigned short* __restrict__ attn) {     // [B,S,1024]
  __shared__ unsigned short ldsK[2][32 * 64];  // 2 x 4 KB
  __shared__ unsigned short ldsV[2][64 * 32];  // 2 x 4 KB
  __shared__ unsigned short ldsP[8][16 * 32];  // 8 KB (per-wave P tile)

  const int lane = threadIdx.x & 31;
  const int wave = threadIdx.x >> 5;
  const int t    = threadIdx.x;
  const int tile = blockIdx.x & 15;            // S/128 = 16 q-tiles
  const int bh   = blockIdx.x >> 4;            // b*H + head
  const int b    = bh >> 4;
  const int hd   = bh & 15;
  const int q0   = tile * 128 + wave * 16;

  const unsigned short* Qp = Qh + ((size_t)bh * S_LEN + q0) * DK;
  const unsigned short* Kp = Kh + (size_t)bh * S_LEN * DK;
  const unsigned short* Vp = Vt + (size_t)bh * DK * S_LEN;
  const int* mrow = mask + (size_t)b * S_LEN * S_LEN;
  unsigned short* lpw = ldsP[wave];

  const int nl = lane & 15, hl = lane >> 4;
  const int kr = t >> 3, kc = (t & 7) * 8;     // K tile: 32 rows x 8 chunks
  const int vr = t >> 2, vc = (t & 3) * 8;     // V tile: 64 rows x 4 chunks

  const v16bf qa0 = load_frag_a(Qp, DK, lane);        // dk [0,32)
  const v16bf qa1 = load_frag_a(Qp + 32, DK, lane);   // dk [32,64)

  const v8f zero = {0.f, 0.f, 0.f, 0.f, 0.f, 0.f, 0.f, 0.f};
  v8f o0 = zero, o1 = zero, o2 = zero, o3 = zero;
  float mr[8], lr[8];
#pragma unroll
  for (int v = 0; v < 8; ++v) { mr[v] = -3.0e38f; lr[v] = 0.f; }

  // prologue: stage key tile 0
  cp16(&ldsK[0][kr * 64 + kc], Kp + (size_t)kr * DK + kc);
  cp16(&ldsV[0][vr * 32 + vc], Vp + (size_t)vr * S_LEN + vc);

  for (int it = 0; it < S_LEN / 32; ++it) {
    const int k0 = it * 32;
    wait_stage();                              // buffers (it&1) ready
    if (it + 1 < S_LEN / 32) {                 // overlap: stage next tiles
      cp16(&ldsK[(it + 1) & 1][kr * 64 + kc],
           Kp + (size_t)(k0 + 32 + kr) * DK + kc);
      cp16(&ldsV[(it + 1) & 1][vr * 32 + vc],
           Vp + (size_t)vr * S_LEN + (k0 + 32) + vc);
    }
    const unsigned short* Kt = ldsK[it & 1];
    const unsigned short* Vl = ldsV[it & 1];

    // ---- S = Q K^T  (16 q x 32 keys), scaled 1/sqrt(64) ----
    v8f s0 = zero, s1 = zero;
    s0 = wmma_bf16(qa0, load_frag_b(Kt, 64, lane), s0);
    s0 = wmma_bf16(qa1, load_frag_b(Kt + 32, 64, lane), s0);
    s1 = wmma_bf16(qa0, load_frag_b(Kt + 16 * 64, 64, lane), s1);
    s1 = wmma_bf16(qa1, load_frag_b(Kt + 16 * 64 + 32, 64, lane), s1);

    float sc0[8], sc1[8];
#pragma unroll
    for (int v = 0; v < 8; ++v) {
      const int qrow = q0 + hl * 8 + v;
      float a0 = s0[v] * 0.125f;
      float a1 = s1[v] * 0.125f;
      a0 = (mrow[(size_t)qrow * S_LEN + k0 + nl] == 0) ? -1.0e9f : a0;
      a1 = (mrow[(size_t)qrow * S_LEN + k0 + 16 + nl] == 0) ? -1.0e9f : a1;
      sc0[v] = a0; sc1[v] = a1;
    }

    // ---- online softmax (rows live across 16 lanes -> width-16 reduce) ----
#pragma unroll
    for (int v = 0; v < 8; ++v) {
      float mx = fmaxf(sc0[v], sc1[v]);
#pragma unroll
      for (int off = 1; off < 16; off <<= 1)
        mx = fmaxf(mx, __shfl_xor(mx, off, 16));
      const float mn = fmaxf(mr[v], mx);
      const float alpha = __expf(mr[v] - mn);
      mr[v] = mn;

      const float p0 = __expf(sc0[v] - mn);
      const float p1 = __expf(sc1[v] - mn);
      float rs = p0 + p1;
#pragma unroll
      for (int off = 1; off < 16; off <<= 1)
        rs += __shfl_xor(rs, off, 16);
      lr[v] = lr[v] * alpha + rs;

      o0[v] *= alpha; o1[v] *= alpha; o2[v] *= alpha; o3[v] *= alpha;
      lpw[(hl * 8 + v) * 32 + nl]      = f2bf(p0);
      lpw[(hl * 8 + v) * 32 + 16 + nl] = f2bf(p1);
    }

    // ---- O += P . V  (P: 16x32 A-frag from LDS; V^T rows contiguous) ----
    const v16bf pa = load_frag_a(lpw, 32, lane);
    o0 = wmma_bf16(pa, load_frag_b(Vl + 0 * 16 * 32, 32, lane), o0);
    o1 = wmma_bf16(pa, load_frag_b(Vl + 1 * 16 * 32, 32, lane), o1);
    o2 = wmma_bf16(pa, load_frag_b(Vl + 2 * 16 * 32, 32, lane), o2);
    o3 = wmma_bf16(pa, load_frag_b(Vl + 3 * 16 * 32, 32, lane), o3);
  }

  // ---- normalize and merge heads: attn[b][q][hd*64 + d] ----
#pragma unroll
  for (int v = 0; v < 8; ++v) {
    const float inv = 1.0f / lr[v];
    const int qrow = q0 + hl * 8 + v;
    unsigned short* dst = attn + ((size_t)(b * S_LEN + qrow)) * DMODEL + hd * DK;
    dst[0 * 16 + nl] = f2bf(o0[v] * inv);
    dst[1 * 16 + nl] = f2bf(o1[v] * inv);
    dst[2 * 16 + nl] = f2bf(o2[v] * inv);
    dst[3 * 16 + nl] = f2bf(o3[v] * inv);
  }
}

// ---------------------------------------------------------------------------
extern "C" void kernel_launch(void* const* d_in, const int* in_sizes, int n_in,
                              void* d_out, int out_size, void* d_ws, size_t ws_size,
                              hipStream_t stream) {
  (void)in_sizes; (void)n_in; (void)out_size; (void)ws_size;

  const float* query = (const float*)d_in[0];
  const float* key_  = (const float*)d_in[1];
  const float* value = (const float*)d_in[2];
  const int*   mask  = (const int*)d_in[3];
  const float* wq = (const float*)d_in[4];
  const float* bq = (const float*)d_in[5];
  const float* wk = (const float*)d_in[6];
  const float* bk = (const float*)d_in[7];
  const float* wv = (const float*)d_in[8];
  const float* bv = (const float*)d_in[9];
  const float* wo = (const float*)d_in[10];
  const float* bo = (const float*)d_in[11];

  // workspace layout (bytes; everything 256-aligned by construction): 64 MB
  const size_t SZ_ACT = (size_t)MROWS * DMODEL * 2;   // 8 MB each
  const size_t SZ_W   = (size_t)DMODEL * DMODEL * 2;  // 2 MB each
  char* ws = (char*)d_ws;
  unsigned short* xq   = (unsigned short*)(ws);
  unsigned short* xk   = (unsigned short*)(ws + SZ_ACT);
  unsigned short* xv   = (unsigned short*)(ws + 2 * SZ_ACT);
  unsigned short* wqb  = (unsigned short*)(ws + 3 * SZ_ACT);
  unsigned short* wkb  = (unsigned short*)(ws + 3 * SZ_ACT + SZ_W);
  unsigned short* wvb  = (unsigned short*)(ws + 3 * SZ_ACT + 2 * SZ_W);
  unsigned short* wob  = (unsigned short*)(ws + 3 * SZ_ACT + 3 * SZ_W);
  unsigned short* Qh   = (unsigned short*)(ws + 3 * SZ_ACT + 4 * SZ_W);
  unsigned short* Kh   = (unsigned short*)(ws + 4 * SZ_ACT + 4 * SZ_W);
  unsigned short* Vt   = (unsigned short*)(ws + 5 * SZ_ACT + 4 * SZ_W);
  unsigned short* attn = (unsigned short*)(ws + 6 * SZ_ACT + 4 * SZ_W);

  // fp32 -> bf16
  const int nAct4 = (MROWS * DMODEL) / 4, nW4 = (DMODEL * DMODEL) / 4;
  cvt_bf16_kernel<<<(nAct4 + 255) / 256, 256, 0, stream>>>(query, xq, nAct4);
  cvt_bf16_kernel<<<(nAct4 + 255) / 256, 256, 0, stream>>>(key_,  xk, nAct4);
  cvt_bf16_kernel<<<(nAct4 + 255) / 256, 256, 0, stream>>>(value, xv, nAct4);
  cvt_bf16_kernel<<<(nW4 + 255) / 256, 256, 0, stream>>>(wq, wqb, nW4);
  cvt_bf16_kernel<<<(nW4 + 255) / 256, 256, 0, stream>>>(wk, wkb, nW4);
  cvt_bf16_kernel<<<(nW4 + 255) / 256, 256, 0, stream>>>(wv, wvb, nW4);
  cvt_bf16_kernel<<<(nW4 + 255) / 256, 256, 0, stream>>>(wo, wob, nW4);

  // Q/K/V projections (WMMA): Q,K -> [B,H,S,dk]; V -> [B,H,dk,S] (transposed)
  dim3 pg(DMODEL / 64, MROWS / 256);
  proj_gemm_kernel<<<pg, 256, 0, stream>>>(xq, wqb, bq, Qh, nullptr, 0);
  proj_gemm_kernel<<<pg, 256, 0, stream>>>(xk, wkb, bk, Kh, nullptr, 0);
  proj_gemm_kernel<<<pg, 256, 0, stream>>>(xv, wvb, bv, Vt, nullptr, 1);

  // flash attention: grid = B*H * (S/128) = 512 blocks
  flash_attn_kernel<<<2 * NHEAD * (S_LEN / 128), 256, 0, stream>>>(
      Qh, Kh, Vt, mask, attn);

  // output projection (fp32 result)
  proj_gemm_kernel<<<pg, 256, 0, stream>>>(attn, wob, bo, nullptr, (float*)d_out, 2);
}
